// InterestDict_71511205478459
// MI455X (gfx1250) — matrix operand order repaired and checked by reading, not compiled, not run on hardware
//
#include <hip/hip_runtime.h>

// ---------------------------------------------------------------------------
// Problem constants (from reference)
// ---------------------------------------------------------------------------
#define N_IN      16384
#define M_DICT    8192
#define DIM       512
#define TOPK      4
#define DECAY     0.95f
#define CHUNKS    8                 // column chunks over M (1024 cols each)
#define COLS_PER_CHUNK 1024
#define NTILES    (COLS_PER_CHUNK / 32)
#define ROWS_PER_BLOCK 128          // 8 waves x 16 rows
#define LDSK      520               // padded halfs per dict row staged in LDS

typedef __attribute__((ext_vector_type(16))) __bf16 v16bf;
typedef __attribute__((ext_vector_type(8)))  __bf16 v8bf;
typedef __attribute__((ext_vector_type(8)))  float  v8f;

// ---------------------------------------------------------------------------
// Small helpers
// ---------------------------------------------------------------------------
static __device__ inline unsigned short f32_to_bf16(float f) {
    unsigned int u = __float_as_uint(f);
    u = u + 0x7FFFu + ((u >> 16) & 1u);   // round-to-nearest-even
    return (unsigned short)(u >> 16);
}

// Async copy of 128 contiguous bytes global -> LDS (8 x b128).
// IOFFSET is added to BOTH the global address and the LDS address
// (CDNA5 ISA), so one base pair serves all 8 ops.
static __device__ inline void async_stage_128(const unsigned short* g,
                                              unsigned ldsoff) {
    asm volatile(
        "global_load_async_to_lds_b128 %0, %1, off\n\t"
        "global_load_async_to_lds_b128 %0, %1, off offset:16\n\t"
        "global_load_async_to_lds_b128 %0, %1, off offset:32\n\t"
        "global_load_async_to_lds_b128 %0, %1, off offset:48\n\t"
        "global_load_async_to_lds_b128 %0, %1, off offset:64\n\t"
        "global_load_async_to_lds_b128 %0, %1, off offset:80\n\t"
        "global_load_async_to_lds_b128 %0, %1, off offset:96\n\t"
        "global_load_async_to_lds_b128 %0, %1, off offset:112"
        :: "v"(ldsoff), "v"(g) : "memory");
}

static __device__ inline void wait_async_le8() {
    asm volatile("s_wait_asynccnt 0x8" ::: "memory");
}
static __device__ inline void wait_async_0() {
    asm volatile("s_wait_asynccnt 0x0" ::: "memory");
}

// ---------------------------------------------------------------------------
// K0: zero a float span (accumulators)
// ---------------------------------------------------------------------------
__global__ void k_zero_f32(float* __restrict__ p, int n) {
    int i = blockIdx.x * blockDim.x + threadIdx.x;
    if (i < n) p[i] = 0.0f;
}

// ---------------------------------------------------------------------------
// K1/K2: fp32 -> bf16 conversion
// ---------------------------------------------------------------------------
__global__ void k_cvt_bf16(const float* __restrict__ in,
                           unsigned short* __restrict__ out, int n) {
    int i = blockIdx.x * blockDim.x + threadIdx.x;
    if (i < n) out[i] = f32_to_bf16(in[i]);
}

// ---------------------------------------------------------------------------
// K3: per-row squared L2 norm of dictionary (fp32). One wave per row.
// ---------------------------------------------------------------------------
__global__ void k_row_norms(const float* __restrict__ dict,
                            float* __restrict__ dnorm) {
    const int lane = threadIdx.x & 31;
    const int wave = threadIdx.x >> 5;
    const int row  = blockIdx.x * 4 + wave;
    const float* dr = dict + (size_t)row * DIM;
    float s = 0.0f;
    #pragma unroll
    for (int j = 0; j < DIM / 32; ++j) {
        float v = dr[lane + 32 * j];
        s += v * v;
    }
    #pragma unroll
    for (int off = 16; off > 0; off >>= 1) s += __shfl_xor(s, off, 32);
    if (lane == 0) dnorm[row] = s;
}

// ---------------------------------------------------------------------------
// K4: fused distance GEMM (bf16 WMMA, f32 accum) + per-row top-4 per chunk.
//   grid = (N/128, CHUNKS), block = 256 (8 waves; each wave owns 16 rows
//   x 32 cols per dict tile via TWO independent accumulator chains).
//   Dict tiles are double-buffered in LDS via async global->LDS loads.
//   score(n,m) = ||d_m||^2 - 2 * x_n . d_m   (||x||^2 dropped: row-constant)
// ---------------------------------------------------------------------------
__global__ __launch_bounds__(256) void k_dist_topk(
    const unsigned short* __restrict__ Xb,     // [N, 512] bf16
    const unsigned short* __restrict__ Db,     // [M, 512] bf16
    const float* __restrict__ dnorm,           // [M]
    float* __restrict__ cand_val,              // [N, CHUNKS, 4]
    int*   __restrict__ cand_idx)              // [N, CHUNKS, 4]
{
    __shared__ unsigned short ldsD[2][32 * LDSK];  // double-buffered dict tile
    __shared__ float          ldsS[8 * 16 * 33];   // per-wave 16x32 score tile

    const int tid  = threadIdx.x;
    const int lane = tid & 31;
    const int wave = tid >> 5;
    const int l16  = lane & 15;
    const int sel  = lane >> 4;                    // half-wave select

    const int row     = blockIdx.x * ROWS_PER_BLOCK + wave * 16 + l16;
    const int c0chunk = blockIdx.y * COLS_PER_CHUNK;

    // ---- Preload all A fragments for this wave's 16-row tile (K = 0..511).
    // A k order used here (contiguous, same as B): lanes 0-15 hold K 0..15 of row=lane,
    // lanes 16-31 hold K 16..31 of row=lane-16.
    const unsigned short* xr = Xb + (size_t)row * DIM;
    v16bf afrag[16];
    #pragma unroll
    for (int kt = 0; kt < 16; ++kt) {
        const int kb = kt * 32;
        v16bf a = *(const v16bf*)(xr + kb + sel * 16);
        afrag[kt] = a;
    }

    // Per-thread staging slice: dict row r of the tile, 128-byte segment seg.
    const int r   = tid >> 3;   // 0..31
    const int seg = tid & 7;    // 0..7

    // Kick off tile 0 into buffer 0.
    async_stage_128(Db + (size_t)(c0chunk + r) * DIM + seg * 64,
                    (unsigned)(size_t)(&ldsD[0][r * LDSK + seg * 64]));

    // Register-resident sorted top-4 (ascending) per row, held in lanes 0-15.
    float tv0 = 3.4e38f, tv1 = 3.4e38f, tv2 = 3.4e38f, tv3 = 3.4e38f;
    int   ti0 = 0,       ti1 = 0,       ti2 = 0,       ti3 = 0;

    int cur = 0;
    #pragma unroll 1
    for (int it = 0; it < NTILES; ++it) {
        const int c0 = c0chunk + it * 32;

        // Issue next tile into the other buffer, then wait for current tile.
        if (it + 1 < NTILES) {
            async_stage_128(Db + (size_t)(c0 + 32 + r) * DIM + seg * 64,
                            (unsigned)(size_t)(&ldsD[cur ^ 1][r * LDSK + seg * 64]));
            if (it + 2 < NTILES) {
                __builtin_prefetch(Db + (size_t)(c0 + 64 + r) * DIM + seg * 64,
                                   0, 1);
            }
            wait_async_le8();   // this wave's tile-it async loads complete
        } else {
            wait_async_0();
        }
        __syncthreads();        // all waves' tile-it data resident in LDS

        // ---- 16x32 output: two independent WMMA chains sharing A fragments.
        v8f acc0 = {};
        v8f acc1 = {};
        // B 32x16 layout: lanes 0-15 = col lane, K kb..kb+15 contiguous;
        // lanes 16-31 = col lane-16, K kb+16..kb+31.
        const unsigned short* base  = &ldsD[cur][0];
        const unsigned short* brow0 = base + (size_t)l16 * LDSK;
        const unsigned short* brow1 = base + (size_t)(16 + l16) * LDSK;
        #pragma unroll
        for (int kt = 0; kt < 16; ++kt) {
            v16bf b0 = *(const v16bf*)(brow0 + kt * 32 + sel * 16);
            v16bf b1 = *(const v16bf*)(brow1 + kt * 32 + sel * 16);
            acc0 = __builtin_amdgcn_wmma_f32_16x16x32_bf16(
                       false, afrag[kt], false, b0, (short)0, acc0, false, false);
            acc1 = __builtin_amdgcn_wmma_f32_16x16x32_bf16(
                       false, afrag[kt], false, b1, (short)0, acc1, false, false);
        }

        // ---- score = ||d||^2 - 2*dot ; transpose via per-wave LDS tile.
        float* st = ldsS + wave * (16 * 33);
        const float dn0 = dnorm[c0 + l16];
        const float dn1 = dnorm[c0 + 16 + l16];
        #pragma unroll
        for (int r8 = 0; r8 < 8; ++r8) {
            st[(r8 + sel * 8) * 33 + l16]      = dn0 - 2.0f * acc0[r8];
            st[(r8 + sel * 8) * 33 + 16 + l16] = dn1 - 2.0f * acc1[r8];
        }
        __builtin_amdgcn_wave_barrier();   // per-wave LDS ops are in-order

        if (lane < 16) {
            #pragma unroll
            for (int cc = 0; cc < 32; ++cc) {
                float v  = st[lane * 33 + cc];
                int   gi = c0 + cc;
                if (v < tv3) {                       // bubble-insert
                    tv3 = v; ti3 = gi;
                    if (tv3 < tv2) {
                        float tf = tv2; tv2 = tv3; tv3 = tf;
                        int   t  = ti2; ti2 = ti3; ti3 = t;
                        if (tv2 < tv1) {
                            tf = tv1; tv1 = tv2; tv2 = tf;
                            t  = ti1; ti1 = ti2; ti2 = t;
                            if (tv1 < tv0) {
                                tf = tv0; tv0 = tv1; tv1 = tf;
                                t  = ti0; ti0 = ti1; ti1 = t;
                            }
                        }
                    }
                }
            }
        }
        __syncthreads();   // all reads of ldsD[cur]/ldsS done before reuse
        cur ^= 1;
    }

    if (lane < 16) {
        const int base2 = (row * CHUNKS + blockIdx.y) * TOPK;
        cand_val[base2 + 0] = tv0; cand_idx[base2 + 0] = ti0;
        cand_val[base2 + 1] = tv1; cand_idx[base2 + 1] = ti1;
        cand_val[base2 + 2] = tv2; cand_idx[base2 + 2] = ti2;
        cand_val[base2 + 3] = tv3; cand_idx[base2 + 3] = ti3;
    }
}

// ---------------------------------------------------------------------------
// K5: merge per-chunk candidates -> global top-4; emit topk_idx + group_emb;
//     scatter-add encodings^T @ X into ws accumulators. One wave per row.
// ---------------------------------------------------------------------------
__global__ void k_merge_scatter(
    const float* __restrict__ cand_val,
    const int*   __restrict__ cand_idx,
    const float* __restrict__ X,
    const float* __restrict__ dict,
    float* __restrict__ out_group,   // [N, 512]
    float* __restrict__ out_idx,     // [N, 4] (indices stored as float)
    float* __restrict__ ws_isum,     // [M, 512]
    float* __restrict__ ws_csize)    // [M]
{
    const int lane = threadIdx.x & 31;
    const int wave = threadIdx.x >> 5;
    const int row  = blockIdx.x * 8 + wave;

    // 32 candidates per row (CHUNKS*TOPK), all distinct indices.
    float v  = cand_val[row * 32 + lane];
    int   gi = cand_idx[row * 32 + lane];

    int topi[TOPK];
    #pragma unroll
    for (int s = 0; s < TOPK; ++s) {
        float bv = v; int bi = gi;
        #pragma unroll
        for (int off = 16; off > 0; off >>= 1) {
            float ov = __shfl_xor(bv, off, 32);
            int   oi = __shfl_xor(bi, off, 32);
            if (ov < bv || (ov == bv && oi < bi)) { bv = ov; bi = oi; }
        }
        topi[s] = bi;
        if (gi == bi) v = 3.4e38f;   // remove winner (indices unique)
    }

    if (lane < TOPK) out_idx[row * TOPK + lane] = (float)topi[lane];

    // group_emb = mean of the 4 nearest codebook rows (fp32 gather).
    float acc[DIM / 32];
    #pragma unroll
    for (int j = 0; j < DIM / 32; ++j) acc[j] = 0.0f;
    #pragma unroll
    for (int s = 0; s < TOPK; ++s) {
        const float* dr = dict + (size_t)topi[s] * DIM;
        #pragma unroll
        for (int j = 0; j < DIM / 32; ++j) acc[j] += dr[lane + 32 * j];
    }
    float* og = out_group + (size_t)row * DIM;
    #pragma unroll
    for (int j = 0; j < DIM / 32; ++j) og[lane + 32 * j] = acc[j] * 0.25f;

    // encodings^T @ X : 4-nnz-per-row sparse GEMM as atomic scatter-add.
    const float* xrow = X + (size_t)row * DIM;
    float xv[DIM / 32];
    #pragma unroll
    for (int j = 0; j < DIM / 32; ++j) xv[j] = xrow[lane + 32 * j];
    #pragma unroll
    for (int s = 0; s < TOPK; ++s) {
        float* is = ws_isum + (size_t)topi[s] * DIM;
        #pragma unroll
        for (int j = 0; j < DIM / 32; ++j) atomicAdd(is + lane + 32 * j, xv[j]);
    }
    if (lane == 0) {
        #pragma unroll
        for (int s = 0; s < TOPK; ++s) atomicAdd(ws_csize + topi[s], 1.0f);
    }
}

// ---------------------------------------------------------------------------
// K6: EMA dictionary update
// ---------------------------------------------------------------------------
__global__ void k_finalize_dict(const float* __restrict__ dict,
                                const float* __restrict__ isum,
                                const float* __restrict__ csize,
                                float* __restrict__ out_dict, int n) {
    int i = blockIdx.x * blockDim.x + threadIdx.x;
    if (i >= n) return;
    int   m  = i >> 9;          // DIM = 512
    float cs = csize[m];
    float dv = dict[i];
    float o  = dv;
    if (cs != 0.0f) {
        float mean = isum[i] / fmaxf(cs, 1.0f);
        o = dv * DECAY + mean * (1.0f - DECAY);
    }
    out_dict[i] = o;
}

// ---------------------------------------------------------------------------
// K7: running sums
// ---------------------------------------------------------------------------
__global__ void k_finalize_sums(const float* __restrict__ css,
                                const float* __restrict__ cun,
                                const float* __restrict__ csize,
                                float* __restrict__ out_css,
                                float* __restrict__ out_cun, int n) {
    int i = blockIdx.x * blockDim.x + threadIdx.x;
    if (i >= n) return;
    float cs = csize[i];
    out_css[i] = css[i] + cs;
    out_cun[i] = cun[i] + ((cs != 0.0f) ? 1.0f : 0.0f);
}

// ---------------------------------------------------------------------------
// Launcher
// ---------------------------------------------------------------------------
extern "C" void kernel_launch(void* const* d_in, const int* in_sizes, int n_in,
                              void* d_out, int out_size, void* d_ws,
                              size_t ws_size, hipStream_t stream) {
    (void)in_sizes; (void)n_in; (void)out_size; (void)ws_size;

    const float* X    = (const float*)d_in[0];   // [16384, 512]
    const float* dict = (const float*)d_in[1];   // [8192, 512]
    const float* css  = (const float*)d_in[2];   // [8192]
    const float* cun  = (const float*)d_in[3];   // [8192]

    // Workspace layout (bytes), 128B-aligned sections, ~44 MB total.
    char* ws = (char*)d_ws;
    unsigned short* Xb       = (unsigned short*)(ws + 0);          // 16 MB
    unsigned short* Db       = (unsigned short*)(ws + 16777216);   //  8 MB
    float*          dnorm    = (float*)(ws + 25165824);            // 32 KB
    float*          cand_val = (float*)(ws + 25198592);            //  2 MB
    int*            cand_idx = (int*)(ws + 27295744);              //  2 MB
    float*          isum     = (float*)(ws + 29392896);            // 16 MB
    float*          csize    = (float*)(ws + 46170112);            // 32 KB
    // NOTE: isum and csize are contiguous -> single zeroing span.

    float* out       = (float*)d_out;
    float* out_group = out;                                  // [N, 512]
    float* out_idx   = out_group + (size_t)N_IN * DIM;       // [N, 4]
    float* out_dict  = out_idx + (size_t)N_IN * TOPK;        // [M, 512]
    float* out_css   = out_dict + (size_t)M_DICT * DIM;      // [M]
    float* out_cun   = out_css + M_DICT;                     // [M]

    const int nz = M_DICT * DIM + M_DICT;
    k_zero_f32<<<(nz + 255) / 256, 256, 0, stream>>>(isum, nz);

    k_cvt_bf16<<<(N_IN * DIM + 255) / 256, 256, 0, stream>>>(X, Xb, N_IN * DIM);
    k_cvt_bf16<<<(M_DICT * DIM + 255) / 256, 256, 0, stream>>>(dict, Db,
                                                               M_DICT * DIM);
    k_row_norms<<<M_DICT / 4, 128, 0, stream>>>(dict, dnorm);

    dim3 g(N_IN / ROWS_PER_BLOCK, CHUNKS);   // (128, 8)
    k_dist_topk<<<g, 256, 0, stream>>>(Xb, Db, dnorm, cand_val, cand_idx);

    k_merge_scatter<<<N_IN / 8, 256, 0, stream>>>(cand_val, cand_idx, X, dict,
                                                  out_group, out_idx, isum,
                                                  csize);

    k_finalize_dict<<<(M_DICT * DIM + 255) / 256, 256, 0, stream>>>(
        dict, isum, csize, out_dict, M_DICT * DIM);
    k_finalize_sums<<<(M_DICT + 255) / 256, 256, 0, stream>>>(
        css, cun, csize, out_css, out_cun, M_DICT);
}
